// MultiheadAttention_10660108829338
// MI455X (gfx1250) — matrix operand order repaired and checked by reading, not compiled, not run on hardware
//
#include <hip/hip_runtime.h>
#include <hip/hip_bf16.h>

// ---------------------------------------------------------------------------
// Causal multi-head attention for MI455X (gfx1250, wave32, WMMA).
//   q,k,v : fp32 [N,L,E] with E=1024, H=16, hd=64, N=2, L=S=2048
//   out   : y [N,L,E] fp32  ++  attn [N,H,L,S] fp32   (concatenated flat)
// Strategy: f16 WMMA (v_wmma_f32_16x16x32_f16) for both GEMMs, f32 softmax,
// full 16 x S score strip held in LDS (128 KB / workgroup, 2 WG per WGP).
// The 512 MB attn output is streamed with nontemporal stores so it does not
// evict the (fully L2-resident, 16 MB) f16 K/V working set.
// ---------------------------------------------------------------------------

#define AN 2
#define AH 16
#define AL 2048
#define AS 2048
#define AE 1024
#define AHD 64
#define SROW 2052              // LDS row stride in floats (2052*4 B = 16B-aligned)
#define SMEM_BYTES (16 * SROW * 4)

typedef __attribute__((ext_vector_type(16))) _Float16 v16h;
typedef __attribute__((ext_vector_type(8)))  _Float16 v8h;
typedef __attribute__((ext_vector_type(8)))  float    v8f;
typedef __attribute__((ext_vector_type(4)))  float    v4f;

union HalfVec {
    v16h v;
    _Float16 h[16];
    struct { v8h lo; v8h hi; } p;
};

// ---------------------------------------------------------------------------
// Pass 1: fp32 -> f16 with per-head layout change.
//   q16,k16: [N,H,L,64] (row-major per head)   v16t: [N,H,64,S] (d-major)
// ---------------------------------------------------------------------------
__global__ void mha_cvt_f16(const float* __restrict__ q,
                            const float* __restrict__ k,
                            const float* __restrict__ v,
                            _Float16* __restrict__ q16,
                            _Float16* __restrict__ k16,
                            _Float16* __restrict__ v16t) {
    size_t i = (size_t)blockIdx.x * blockDim.x + threadIdx.x;
    if (i >= (size_t)AN * AL * AE) return;
    int e = (int)(i % AE);
    size_t nl = i / AE;
    int l = (int)(nl % AL);
    int n = (int)(nl / AL);
    int h = e >> 6;
    int d = e & 63;
    size_t hb = (size_t)n * AH + h;
    q16[(hb * AL + l) * AHD + d] = (_Float16)q[i];
    k16[(hb * AS + l) * AHD + d] = (_Float16)k[i];
    v16t[(hb * AHD + d) * AS + l] = (_Float16)v[i];
}

// ---------------------------------------------------------------------------
// Pass 2: one workgroup (4 waves / 128 threads) per (n, h, 16-row tile).
// ---------------------------------------------------------------------------
__global__ void __launch_bounds__(128)
mha_attn_kernel(const _Float16* __restrict__ q16,
                const _Float16* __restrict__ k16,
                const _Float16* __restrict__ v16t,
                float* __restrict__ yOut,
                float* __restrict__ attnOut) {
    extern __shared__ float sS[];   // [16][SROW] f32 score / prob strip

    const int tile = blockIdx.x;
    const int l0 = (tile & ((AL / 16) - 1)) * 16;
    const int h  = (tile / (AL / 16)) & (AH - 1);
    const int n  = tile / ((AL / 16) * AH);

    const int lane = threadIdx.x & 31;
    const int wave = threadIdx.x >> 5;
    const int rowA = lane & 15;       // A row / B col / C col for this lane
    const int kgrp = lane >> 4;       // lane group (ISA 7.12.2 layouts)
    const int kbaseA = kgrp * 8;      // unused; A and B both load at kbaseB
    const int kbaseB = kgrp * 16;     // K offset for A and B (this code's contiguous k = 16*kgrp + s)

    const size_t headQ = ((size_t)n * AH + h) * AL;   // rows of q16 (per head)
    const size_t headK = ((size_t)n * AH + h) * AS;   // rows of k16
    const size_t headV = ((size_t)n * AH + h) * AHD;  // rows of v16t
    const float scale = 0.125f;                        // 1/sqrt(64)

    // ---- load Q fragments once per wave (d = 64 -> two K=32 chunks) -------
    HalfVec qa[2];
    {
        const _Float16* qp = q16 + (headQ + (l0 + rowA)) * AHD;
        for (int kc = 0; kc < 2; ++kc) {
            qa[kc].p.lo = *(const v8h*)(qp + kc * 32 + kbaseB);
            qa[kc].p.hi = *(const v8h*)(qp + kc * 32 + kbaseB + 8);
        }
    }

    // ---- phase 1: causal scores S = (Q K^T) * scale into LDS --------------
    const int nsc = (l0 >> 4) + 1;            // 16-wide s-chunks this tile needs
    for (int sc = wave; sc < nsc; sc += 4) {
        const int s0 = sc * 16;
        if (sc + 4 < nsc) {                    // prefetch next K tile this wave
            __builtin_prefetch(k16 + (headK + (s0 + 64 + rowA)) * AHD, 0, 3);
        }
        v8f acc = {};
        for (int kc = 0; kc < 2; ++kc) {
            const _Float16* kp =
                k16 + (headK + (s0 + rowA)) * AHD + kc * 32 + kbaseB;
            v16h b = *(const v16h*)kp;
            acc = __builtin_amdgcn_wmma_f32_16x16x32_f16(
                false, qa[kc].v, false, b, (short)0, acc, false, false);
        }
        const int colS = s0 + rowA;
        for (int r = 0; r < 8; ++r) {
            const int row = r + (kgrp << 3);
            const float val =
                (colS <= l0 + row) ? acc[r] * scale : -__builtin_inff();
            sS[row * SROW + colS] = val;
        }
    }
    __syncthreads();

    // ---- phase 2: row softmax; write P to LDS + global attn (NT stores) ---
    const int Send = l0 + 16;                  // columns with computed scores
    const int nch32 = (Send + 31) >> 5;        // K=32 chunks for the PV GEMM
    const int Spad = nch32 * 32;               // zero-padded LDS extent
    for (int rr = 0; rr < 4; ++rr) {
        const int row = wave * 4 + rr;
        const int l = l0 + row;
        float* rowp = &sS[row * SROW];

        // pass 1: row max (float4 / ds_load_b128; Send is a multiple of 16)
        float m = -__builtin_inff();
        for (int s = lane << 2; s < Send; s += 128) {
            v4f t = *(const v4f*)(rowp + s);
            m = fmaxf(m, fmaxf(fmaxf(t.x, t.y), fmaxf(t.z, t.w)));
        }
        for (int off = 16; off > 0; off >>= 1)
            m = fmaxf(m, __shfl_xor(m, off, 32));

        // pass 2: e = exp(x - m) once, written back to LDS; accumulate sum
        float sum = 0.0f;
        for (int s = lane << 2; s < Send; s += 128) {
            v4f t = *(const v4f*)(rowp + s);     // masked -> -inf -> exp = 0
            t.x = __expf(t.x - m);
            t.y = __expf(t.y - m);
            t.z = __expf(t.z - m);
            t.w = __expf(t.w - m);
            *(v4f*)(rowp + s) = t;
            sum += (t.x + t.y) + (t.z + t.w);
        }
        for (int off = 16; off > 0; off >>= 1) sum += __shfl_xor(sum, off, 32);
        const float inv = 1.0f / sum;

        // pass 3: normalize in LDS, stream to global attn (nontemporal)
        float* arow = attnOut + (((size_t)n * AH + h) * AL + l) * (size_t)AS;
        for (int s = lane << 2; s < Send; s += 128) {
            v4f t = *(const v4f*)(rowp + s);
            t *= inv;
            *(v4f*)(rowp + s) = t;
            __builtin_nontemporal_store(t, (v4f*)(arow + s));
        }
        for (int s = Send + lane; s < Spad; s += 32) rowp[s] = 0.0f;
        // bulk zero-fill of the causal-masked tail (NT b128 stores)
        const v4f z = {0.0f, 0.0f, 0.0f, 0.0f};
        for (int s = Send + (lane << 2); s < AS; s += 32 * 4)
            __builtin_nontemporal_store(z, (v4f*)(arow + s));
    }
    __syncthreads();

    // ---- phase 3: Y = P V ; each wave owns a 16-wide slice of hd=64 -------
    const int d0 = wave * 16;
    v8f yacc = {};
    for (int sc = 0; sc < nch32; ++sc) {
        HalfVec pa;
        const float* pr = &sS[rowA * SROW + sc * 32 + kbaseB];
        for (int i = 0; i < 8; ++i) {
            pa.h[i]     = (_Float16)pr[i];
            pa.h[8 + i] = (_Float16)pr[8 + i];
        }
        const _Float16* vp =
            v16t + (headV + (d0 + rowA)) * AS + sc * 32 + kbaseB;
        v16h b = *(const v16h*)vp;
        yacc = __builtin_amdgcn_wmma_f32_16x16x32_f16(
            false, pa.v, false, b, (short)0, yacc, false, false);
    }
    for (int r = 0; r < 8; ++r) {
        const int row = r + (kgrp << 3);
        yOut[((size_t)n * AL + (l0 + row)) * AE + h * AHD + d0 + rowA] = yacc[r];
    }
}

// ---------------------------------------------------------------------------
extern "C" void kernel_launch(void* const* d_in, const int* in_sizes, int n_in,
                              void* d_out, int out_size, void* d_ws, size_t ws_size,
                              hipStream_t stream) {
    const float* q = (const float*)d_in[0];
    const float* k = (const float*)d_in[1];
    const float* v = (const float*)d_in[2];
    float* out = (float*)d_out;

    const size_t headElems = (size_t)AN * AH * AL * AHD;   // 4 Mi elements each
    _Float16* q16  = (_Float16*)d_ws;
    _Float16* k16  = q16 + headElems;
    _Float16* v16t = k16 + headElems;

    const size_t yElems = (size_t)AN * AL * AE;
    float* yOut = out;
    float* attnOut = out + yElems;

    const size_t cvtTotal = (size_t)AN * AL * AE;
    mha_cvt_f16<<<dim3((unsigned)((cvtTotal + 255) / 256)), dim3(256), 0, stream>>>(
        q, k, v, q16, k16, v16t);

    const int tiles = AN * AH * (AL / 16);   // 4096 workgroups
    mha_attn_kernel<<<dim3(tiles), dim3(128), SMEM_BYTES, stream>>>(
        q16, k16, v16t, yOut, attnOut);
}
